// HierarchicalMambaBackbone_29686813950061
// MI455X (gfx1250) — matrix operand order repaired and checked by reading, not compiled, not run on hardware
//
#include <hip/hip_runtime.h>
#include <hip/hip_bf16.h>

// ---------------------------------------------------------------------------
// CDNA5 / gfx1250 implementation of the hierarchical bidirectional-Mamba
// vision backbone. All dense projections run through v_wmma_f32_16x16x32_bf16
// with LDS-staged bf16 tiles (ds_load_b128 fragment reads), vectorized b128
// tile staging on interior tiles, branchless edge handling, and residual/
// activation fused into the GEMM epilogue. wave32 throughout.
// ---------------------------------------------------------------------------

typedef __attribute__((ext_vector_type(16))) __bf16          v16bf;
typedef __attribute__((ext_vector_type(8)))  float           v8f;
typedef __attribute__((ext_vector_type(8)))  unsigned short  u16x8;
typedef __attribute__((ext_vector_type(4)))  unsigned short  u16x4;
typedef __attribute__((ext_vector_type(4)))  float           f32x4;

union BFrag { v16bf v; unsigned short u[16]; u16x8 q[2]; };

__device__ __forceinline__ unsigned short f32_to_bf16(float f) {
  union { float f; unsigned u; } c; c.f = f;
  unsigned u = c.u;
  unsigned r = u + 0x7FFFu + ((u >> 16) & 1u);   // round-to-nearest-even
  return (unsigned short)(r >> 16);
}

__device__ __forceinline__ float act_apply(float v, int act) {
  if (act == 1) {                       // gelu (tanh approx, jax default)
    const float c = 0.7978845608028654f;
    float t = tanhf(c * (v + 0.044715f * v * v * v));
    return 0.5f * v * (1.0f + t);
  } else if (act == 2) {                // silu
    return v / (1.0f + __expf(-v));
  } else if (act == 3) {                // softplus (stable)
    float ax = fabsf(v);
    return log1pf(__expf(-ax)) + fmaxf(v, 0.0f);
  }
  return v;
}

// ---------------------------------------------------------------------------
// WMMA GEMM: C[M,N] = [C +] act(A[M,K] @ B[K,N] + bias)
// Block = 128 threads = 4 waves = 2x2 grid of 16x16 C tiles (32x32 per block).
// K chunk = 64 -> two v_wmma per chunk. A/B staged in LDS as bf16; B stored
// transposed [n][k] so every fragment read is a contiguous ds_load_b128.
// Interior tiles with 16B-aligned rows use float4 staging (global_load_b128);
// ragged tiles use index-clamp + value-select (no EXEC divergence anywhere
// near the WMMAs, as the ISA requires EXEC==all-ones).
// ---------------------------------------------------------------------------
#define TM 32
#define TN 32
#define TK 64
#define LDSP (TK + 8)   // padded row: 72 halves = 144 B (16B-aligned stride)

__global__ __launch_bounds__(128)
void k_gemm(const float* __restrict__ A, int lda,
            const float* __restrict__ B, int ldb,
            const float* __restrict__ bias,
            float* __restrict__ C, int ldc,
            int M, int N, int K, int act, int accum) {
  __shared__ alignas(16) unsigned short As[TM][LDSP];
  __shared__ alignas(16) unsigned short Bs[TN][LDSP];

  const int tid  = threadIdx.x;
  const int lane = tid & 31;
  const int wave = tid >> 5;
  const int wm = wave >> 1;            // wave M quadrant (0..1)
  const int wn = wave & 1;             // wave N quadrant (0..1)
  const int tm = blockIdx.y * TM;
  const int tn = blockIdx.x * TN;
  const int hi = (lane >> 4) & 1;      // lane half-group
  const int lm = lane & 15;

  // uniform per-matrix fast-path eligibility (row alignment)
  const bool alnA = ((lda & 3) == 0) && (tm + TM <= M);
  const bool alnB = ((ldb & 3) == 0) && (tn + TN <= N);

  v8f acc = {};
  for (int k0 = 0; k0 < K; k0 += TK) {
    const bool kin = (k0 + TK <= K);
    // ---- stage A tile (TM x TK) ----
    if (alnA && kin) {
      // 512 float4 slots, 4 per thread: vectorized global_load_b128
#pragma unroll
      for (int i = 0; i < 4; ++i) {
        int s  = tid + 128 * i;
        int ml = s >> 4;               // 16 float4 per 64-wide row
        int kl = (s & 15) * 4;
        f32x4 v = *(const f32x4*)(A + (size_t)(tm + ml) * lda + (k0 + kl));
        u16x4 p;
        p.x = f32_to_bf16(v.x); p.y = f32_to_bf16(v.y);
        p.z = f32_to_bf16(v.z); p.w = f32_to_bf16(v.w);
        *(u16x4*)&As[ml][kl] = p;      // aligned ds_store_b64
      }
    } else {
      for (int e = tid; e < TM * TK; e += 128) {
        int ml = e >> 6;
        int kl = e & (TK - 1);
        int gm = tm + ml, gk = k0 + kl;
        float v = A[(size_t)min(gm, M - 1) * lda + min(gk, K - 1)];
        As[ml][kl] = (gm < M && gk < K) ? f32_to_bf16(v) : (unsigned short)0;
      }
    }
    // ---- stage B tile (TK x TN) transposed to [n][k] ----
    if (alnB && kin) {
#pragma unroll
      for (int i = 0; i < 4; ++i) {
        int s  = tid + 128 * i;
        int kl = s >> 3;               // 8 float4 per 32-wide row
        int nl = (s & 7) * 4;
        f32x4 v = *(const f32x4*)(B + (size_t)(k0 + kl) * ldb + (tn + nl));
        Bs[nl + 0][kl] = f32_to_bf16(v.x);
        Bs[nl + 1][kl] = f32_to_bf16(v.y);
        Bs[nl + 2][kl] = f32_to_bf16(v.z);
        Bs[nl + 3][kl] = f32_to_bf16(v.w);
      }
    } else {
      for (int e = tid; e < TK * TN; e += 128) {
        int kl = e >> 5;
        int nl = e & (TN - 1);         // coalesced over n
        int gk = k0 + kl, gn = tn + nl;
        float v = B[(size_t)min(gk, K - 1) * ldb + min(gn, N - 1)];
        Bs[nl][kl] = (gk < K && gn < N) ? f32_to_bf16(v) : (unsigned short)0;
      }
    }
    // speculative prefetch of next K tile (lowers to global_prefetch_b8)
    if (k0 + TK < K) {
      __builtin_prefetch(&A[(size_t)min(tm + (tid >> 2), M - 1) * lda +
                            min(k0 + TK + (tid & 3) * 16, K - 1)], 0, 1);
      __builtin_prefetch(&B[(size_t)min(k0 + TK + (tid >> 2), K - 1) * ldb +
                            min(tn + (tid & 3) * 8, N - 1)], 0, 1);
    }
    __syncthreads();

    const unsigned short* arow = &As[wm * 16 + lm][0];
    const unsigned short* brow = &Bs[wn * 16 + lm][0];
#pragma unroll
    for (int kk = 0; kk < TK; kk += 32) {
      BFrag a, b;
      // A frag (16x32): halves i<8 -> k = kk + 8*hi + i ; i>=8 -> +16
      a.q[0] = *(const u16x8*)(arow + kk + 8 * hi);
      a.q[1] = *(const u16x8*)(arow + kk + 16 + 8 * hi);
      // B frag (32x16): halves i<8 -> k = kk + 8*hi + i ; i>=8 -> +16
      b.q[0] = *(const u16x8*)(brow + kk + 8 * hi);
      b.q[1] = *(const u16x8*)(brow + kk + 16 + 8 * hi);
      acc = __builtin_amdgcn_wmma_f32_16x16x32_bf16(
          false, a.v, false, b.v, (short)0, acc, false, false);
    }
    __syncthreads();
  }

  // ---- epilogue: bias + activation + optional residual accumulate ----
  const int n = tn + wn * 16 + lm;
#pragma unroll
  for (int j = 0; j < 8; ++j) {
    int m = tm + wm * 16 + j + hi * 8;
    if (m < M && n < N) {
      float v = acc[j];
      if (bias) v += bias[n];
      v = act_apply(v, act);
      float* cp = &C[(size_t)m * ldc + n];
      *cp = (accum ? *cp : 0.0f) + v;
    }
  }
}

// ---------------------------------------------------------------------------
// LayerNorm over last dim D, one workgroup per row, strided in/out.
// ---------------------------------------------------------------------------
__global__ __launch_bounds__(128)
void k_layernorm(const float* __restrict__ in, int ldi,
                 const float* __restrict__ g, const float* __restrict__ b,
                 float* __restrict__ out, int ldo, int D) {
  __shared__ float s1[128], s2[128];
  const size_t row = blockIdx.x;
  const float* ip = in + row * (size_t)ldi;
  float sum = 0.f, sq = 0.f;
  for (int c = threadIdx.x; c < D; c += 128) {
    float v = ip[c]; sum += v; sq += v * v;
  }
  s1[threadIdx.x] = sum; s2[threadIdx.x] = sq;
  __syncthreads();
  for (int st = 64; st > 0; st >>= 1) {
    if ((int)threadIdx.x < st) {
      s1[threadIdx.x] += s1[threadIdx.x + st];
      s2[threadIdx.x] += s2[threadIdx.x + st];
    }
    __syncthreads();
  }
  float m  = s1[0] / (float)D;
  float vv = s2[0] / (float)D - m * m;
  float rs = rsqrtf(vv + 1e-5f);
  float* op = out + row * (size_t)ldo;
  for (int c = threadIdx.x; c < D; c += 128)
    op[c] = (ip[c] - m) * rs * g[c] + b[c];
}

// Causal depthwise conv1d (kernel 4, left pad 3) + SiLU.
// Input = xi columns of xz (row stride ldx = 2*di); output contiguous [rows,di].
__global__ void k_dwconv_silu(const float* __restrict__ xz, int ldx,
                              const float* __restrict__ w,
                              const float* __restrict__ bias,
                              float* __restrict__ xc, int L, int di, int total) {
  int t = blockIdx.x * blockDim.x + threadIdx.x;
  if (t >= total) return;
  int d = t % di;
  int r = t / di;
  int l = r % L;
  int b = r / L;
  float acc = bias[d];
#pragma unroll
  for (int k = 0; k < 4; ++k) {
    int ls = l - 3 + k;
    if (ls >= 0)
      acc += w[d * 4 + k] * xz[((size_t)b * L + ls) * ldx + d];
  }
  xc[(size_t)r * di + d] = acc / (1.0f + __expf(-acc));
}

// A = -exp(A_log)
__global__ void k_negexp(const float* __restrict__ a_log, float* __restrict__ A,
                         int total) {
  int t = blockIdx.x * blockDim.x + threadIdx.x;
  if (t < total) A[t] = -__expf(a_log[t]);
}

// Selective scan: one thread per (batch, channel); 16-wide state in registers.
__global__ void k_scan(const float* __restrict__ u, const float* __restrict__ dt,
                       const float* __restrict__ A,
                       const float* __restrict__ Bt, int ldbc,
                       const float* __restrict__ Ct,
                       const float* __restrict__ Dp,
                       float* __restrict__ y, int Bb, int L, int di) {
  int t = blockIdx.x * blockDim.x + threadIdx.x;
  if (t >= Bb * di) return;
  int b = t / di, d = t - b * di;
  float Ar[16], h[16];
#pragma unroll
  for (int n = 0; n < 16; ++n) { Ar[n] = A[d * 16 + n]; h[n] = 0.f; }
  const float Dv = Dp[d];
  const size_t base = (size_t)b * L;
  for (int l = 0; l < L; ++l) {
    size_t r = base + l;
    float uv  = u[r * di + d];
    float dtv = dt[r * di + d];
    float du  = dtv * uv;
    const float* Bp = Bt + r * ldbc;
    const float* Cp = Ct + r * ldbc;
    float acc = 0.f;
#pragma unroll
    for (int n = 0; n < 16; ++n) {
      h[n] = __expf(dtv * Ar[n]) * h[n] + du * Bp[n];
      acc += h[n] * Cp[n];
    }
    y[r * di + d] = acc + uv * Dv;
  }
}

// yz = y * silu(z), z = xz[:, di + d] (row stride ld2)
__global__ void k_gate(const float* __restrict__ y, const float* __restrict__ xz,
                       int ld2, int di, float* __restrict__ yz, int total) {
  int t = blockIdx.x * blockDim.x + threadIdx.x;
  if (t >= total) return;
  int d = t % di;
  size_t r = (size_t)(t / di);
  float z = xz[r * ld2 + di + d];
  yz[t] = y[t] * (z / (1.0f + __expf(-z)));
}

// flip along sequence dim: out[b, l, c] = in[b, L-1-l, c]
__global__ void k_flip(const float* __restrict__ in, float* __restrict__ out,
                       int L, int C, int total) {
  int t = blockIdx.x * blockDim.x + threadIdx.x;
  if (t >= total) return;
  int c = t % C;
  int r = t / C;
  int l = r % L;
  int b = r / L;
  out[((size_t)b * L + (L - 1 - l)) * C + c] = in[t];
}

// patch merge rearrange: [B,H,W,C] -> [B,(H/2)(W/2),4C], quadrant order
// (0,0),(1,0),(0,1),(1,1)
__global__ void k_merge(const float* __restrict__ in, float* __restrict__ out,
                        int H, int W, int C, int total) {
  int t = blockIdx.x * blockDim.x + threadIdx.x;
  if (t >= total) return;
  int C4 = 4 * C;
  int c4 = t % C4;
  int p  = t / C4;
  int q  = c4 / C;
  int c  = c4 - q * C;
  int W2 = W / 2;
  int j  = p % W2;
  int rest = p / W2;
  int i  = rest % (H / 2);
  int b  = rest / (H / 2);
  int yy = 2 * i + ((q == 1 || q == 3) ? 1 : 0);
  int xx = 2 * j + ((q >= 2) ? 1 : 0);
  out[t] = in[((size_t)(b * H + yy) * W + xx) * C + c];
}

// strided patchify conv (kernel s, stride s, VALID), NCHW in -> NHWC out
__global__ void k_embed_conv(const float* __restrict__ x,
                             const float* __restrict__ w,
                             const float* __restrict__ bias,
                             float* __restrict__ out, int s, int oh, int total) {
  int t = blockIdx.x * blockDim.x + threadIdx.x;
  if (t >= total) return;
  int c = t % 96;
  int p = t / 96;
  int ox = p % oh; p /= oh;
  int oy = p % oh;
  int b  = p / oh;
  float acc = bias[c];
  for (int ci = 0; ci < 3; ++ci)
    for (int ky = 0; ky < s; ++ky)
      for (int kx = 0; kx < s; ++kx)
        acc += x[(((size_t)b * 3 + ci) * 224 + oy * s + ky) * 224 + ox * s + kx]
             * w[(((size_t)c * 3 + ci) * s + ky) * s + kx];
  out[t] = acc;
}

// bilinear resize NHWC (half-pixel centers), ih x ih -> 56 x 56, C=96
__global__ void k_resize(const float* __restrict__ in, int ih,
                         float* __restrict__ out, int total) {
  int t = blockIdx.x * blockDim.x + threadIdx.x;
  if (t >= total) return;
  int c = t % 96;
  int p = t / 96;
  int ox = p % 56; p /= 56;
  int oy = p % 56;
  int b  = p / 56;
  float sc = (float)ih / 56.0f;
  float sy = (oy + 0.5f) * sc - 0.5f;
  float sx = (ox + 0.5f) * sc - 0.5f;
  int y0 = (int)floorf(sy), x0 = (int)floorf(sx);
  float fy = sy - y0, fx = sx - x0;
  int y1 = y0 + 1, x1 = x0 + 1;
  y0 = min(max(y0, 0), ih - 1); y1 = min(max(y1, 0), ih - 1);
  x0 = min(max(x0, 0), ih - 1); x1 = min(max(x1, 0), ih - 1);
  const float* ib = in + (size_t)b * ih * ih * 96;
  float v00 = ib[((size_t)y0 * ih + x0) * 96 + c];
  float v01 = ib[((size_t)y0 * ih + x1) * 96 + c];
  float v10 = ib[((size_t)y1 * ih + x0) * 96 + c];
  float v11 = ib[((size_t)y1 * ih + x1) * 96 + c];
  float v0 = v00 + (v01 - v00) * fx;
  float v1 = v10 + (v11 - v10) * fx;
  out[t] = v0 + (v1 - v0) * fy;
}

// h[b,l,c] += pos[l,c]  (pos broadcast over batch)
__global__ void k_add_pos(float* __restrict__ h, const float* __restrict__ pos,
                          int per_batch, int total) {
  int t = blockIdx.x * blockDim.x + threadIdx.x;
  if (t < total) h[t] += pos[t % per_batch];
}

// ---------------------------------------------------------------------------
// Host orchestration
// ---------------------------------------------------------------------------
extern "C" void kernel_launch(void* const* d_in, const int* in_sizes, int n_in,
                              void* d_out, int out_size, void* d_ws, size_t ws_size,
                              hipStream_t stream) {
  (void)in_sizes; (void)n_in; (void)out_size; (void)ws_size;
  const float* x   = (const float*)d_in[0];
  const float* pos = (const float*)d_in[1];
  float* outF = (float*)d_out;

  // ---- parameter walk (matches setup_inputs() flatten order) ----
  int idx = 2;
  auto next = [&]() { return (const float*)d_in[idx++]; };

  const float *cw[3], *cb[3], *ng[3], *nb[3];
  for (int i = 0; i < 3; ++i) { cw[i] = next(); cb[i] = next(); }
  for (int i = 0; i < 3; ++i) { ng[i] = next(); nb[i] = next(); }
  const float *fw1 = next(), *fb1 = next(), *fw2 = next(), *fb2 = next();
  const float *fg = next(), *fbb = next();

  struct MambaP { const float *in_proj,*conv_w,*conv_b,*x_proj,*dt_w,*dt_b,*A_log,*Dp,*out_proj; };
  struct LayerP { const float *ln1_g,*ln1_b; MambaP fwd,bwd;
                  const float *ln2_g,*ln2_b,*w1,*b1,*w2,*b2; };
  auto nextM = [&]() {
    MambaP m;
    m.in_proj = next(); m.conv_w = next(); m.conv_b = next(); m.x_proj = next();
    m.dt_w = next(); m.dt_b = next(); m.A_log = next(); m.Dp = next();
    m.out_proj = next();
    return m;
  };
  auto nextL = [&]() {
    LayerP l;
    l.ln1_g = next(); l.ln1_b = next();
    l.fwd = nextM(); l.bwd = nextM();
    l.ln2_g = next(); l.ln2_b = next();
    l.w1 = next(); l.b1 = next(); l.w2 = next(); l.b2 = next();
    return l;
  };
  LayerP layers0[2] = { nextL(), nextL() };
  const float *m0g = next(), *m0b = next(), *m0w = next();
  LayerP layers1[2] = { nextL(), nextL() };
  const float *m1g = next(), *m1b = next(), *m1w = next();
  LayerP layers2[4] = { nextL(), nextL(), nextL(), nextL() };
  const float *fng = next(), *fnb = next();

  // ---- workspace carve (floats) ----
  const size_t R0 = 50176;          // 16 * 56 * 56
  float* Wp = (float*)d_ws;
  size_t off = 0;
  auto alloc = [&](size_t n) { float* p = Wp + off; off += n; return p; };
  float* h    = alloc(R0 * 96);     // residual stream (max rows*D)
  float* hln  = alloc(R0 * 96);     // LN output / h2
  float* hflip= alloc(R0 * 96);     // flipped LN input
  float* xz   = alloc(R0 * 384);    // in_proj out (also hcat / merge rearrange)
  float* xc   = alloc(R0 * 192);    // conv+silu out (also fusion h1)
  float* dbl  = alloc(R0 * 38);     // x_proj out (also 14x14 conv buf)
  float* dtb  = alloc(R0 * 192);    // dt (also 28x28 conv buf)
  float* yb   = alloc(R0 * 192);    // scan out / flip temp (also 56x56 conv buf)
  float* yzb  = alloc(R0 * 192);    // gated (also resize buf)
  float* mlp1 = alloc(R0 * 384);    // MLP hidden (also merge LN out)
  float* Aneg = alloc(768 * 16);

  auto gemm = [&](const float* A, int lda, const float* Bm, int ldb,
                  const float* bias, float* C, int ldc,
                  int M, int N, int K, int act, int accum) {
    dim3 grid((unsigned)((N + TN - 1) / TN), (unsigned)((M + TM - 1) / TM));
    k_gemm<<<grid, 128, 0, stream>>>(A, lda, Bm, ldb, bias, C, ldc,
                                     M, N, K, act, accum);
  };
  auto ln = [&](const float* in, int ldi, const float* g, const float* b,
                float* out, int ldo, int rows, int D) {
    k_layernorm<<<dim3((unsigned)rows), 128, 0, stream>>>(in, ldi, g, b, out, ldo, D);
  };
  auto blk = [](int total) { return dim3((unsigned)((total + 255) / 256)); };

  // ---- bidirectional mamba branch; accumulates out_proj into `target` ----
  auto run_mamba = [&](const float* hin, float* target, int rows, int L,
                       int D, int di, int R, const MambaP& mp, bool flip_out) {
    gemm(hin, D, mp.in_proj, 2 * di, nullptr, xz, 2 * di, rows, 2 * di, D, 0, 0);
    int tci = rows * di;
    k_dwconv_silu<<<blk(tci), 256, 0, stream>>>(xz, 2 * di, mp.conv_w, mp.conv_b,
                                                xc, L, di, tci);
    gemm(xc, di, mp.x_proj, R + 32, nullptr, dbl, R + 32, rows, R + 32, di, 0, 0);
    gemm(dbl, R + 32, mp.dt_w, di, mp.dt_b, dtb, di, rows, di, R, 3 /*softplus*/, 0);
    k_negexp<<<blk(di * 16), 256, 0, stream>>>(mp.A_log, Aneg, di * 16);
    k_scan<<<blk(16 * di), 256, 0, stream>>>(xc, dtb, Aneg, dbl + R, R + 32,
                                             dbl + R + 16, mp.Dp, yb, 16, L, di);
    k_gate<<<blk(tci), 256, 0, stream>>>(yb, xz, 2 * di, di, yzb, tci);
    const float* gsrc = yzb;
    if (flip_out) {
      k_flip<<<blk(tci), 256, 0, stream>>>(yzb, yb, L, di, tci);
      gsrc = yb;
    }
    gemm(gsrc, di, mp.out_proj, D, nullptr, target, D, rows, D, di, 0, 1 /*accum*/);
  };

  auto run_layer = [&](float* hbuf, int rows, int L, int D, const LayerP& lp) {
    int di = 2 * D, R = (D + 15) / 16;
    int thd = rows * D;
    ln(hbuf, D, lp.ln1_g, lp.ln1_b, hln, D, rows, D);
    run_mamba(hln, hbuf, rows, L, D, di, R, lp.fwd, false);
    k_flip<<<blk(thd), 256, 0, stream>>>(hln, hflip, L, D, thd);
    run_mamba(hflip, hbuf, rows, L, D, di, R, lp.bwd, true);
    ln(hbuf, D, lp.ln2_g, lp.ln2_b, hln, D, rows, D);   // h2 reuses hln
    gemm(hln, D, lp.w1, 4 * D, lp.b1, mlp1, 4 * D, rows, 4 * D, D, 1 /*gelu*/, 0);
    gemm(mlp1, 4 * D, lp.w2, D, lp.b2, hbuf, D, rows, D, 4 * D, 0, 1 /*accum*/);
  };

  // ================= patch embed =================
  {
    float* hcat = xz;                                   // [R0, 288]
    // scale 4 -> 56x56 direct
    int t4 = 16 * 56 * 56 * 96;
    k_embed_conv<<<blk(t4), 256, 0, stream>>>(x, cw[0], cb[0], yb, 4, 56, t4);
    ln(yb, 96, ng[0], nb[0], hcat + 0, 288, (int)R0, 96);
    // scale 8 -> 28x28 -> resize 56
    int t8 = 16 * 28 * 28 * 96;
    k_embed_conv<<<blk(t8), 256, 0, stream>>>(x, cw[1], cb[1], dtb, 8, 28, t8);
    k_resize<<<blk(t4), 256, 0, stream>>>(dtb, 28, yzb, t4);
    ln(yzb, 96, ng[1], nb[1], hcat + 96, 288, (int)R0, 96);
    // scale 16 -> 14x14 -> resize 56
    int t16 = 16 * 14 * 14 * 96;
    k_embed_conv<<<blk(t16), 256, 0, stream>>>(x, cw[2], cb[2], dbl, 16, 14, t16);
    k_resize<<<blk(t4), 256, 0, stream>>>(dbl, 14, yzb, t4);
    ln(yzb, 96, ng[2], nb[2], hcat + 192, 288, (int)R0, 96);
    // fusion MLP + LN + pos
    gemm(hcat, 288, fw1, 192, fb1, xc, 192, (int)R0, 192, 288, 1 /*gelu*/, 0);
    gemm(xc, 192, fw2, 96, fb2, hln, 96, (int)R0, 96, 192, 0, 0);
    ln(hln, 96, fg, fbb, h, 96, (int)R0, 96);
    k_add_pos<<<blk(t4), 256, 0, stream>>>(h, pos, 3136 * 96, t4);
  }

  // output layout: out(16*196*384) | f0(16*784*192) | f1(16*196*384) | f2(...)
  const size_t OFF_F0 = 1204224, OFF_F1 = 3612672, OFF_F2 = 4816896;

  int H = 56, W = 56, rows = (int)R0, D = 96;

  // ================= stage 0 =================
  for (int i = 0; i < 2; ++i) run_layer(h, rows, H * W, D, layers0[i]);
  {
    int tmge = 16 * (H / 2) * (W / 2) * 4 * D;
    k_merge<<<blk(tmge), 256, 0, stream>>>(h, xz, H, W, D, tmge);
    ln(xz, 4 * D, m0g, m0b, mlp1, 4 * D, rows / 4, 4 * D);
    gemm(mlp1, 4 * D, m0w, 192, nullptr, h, 192, rows / 4, 192, 4 * D, 0, 0);
    rows /= 4; H /= 2; W /= 2; D = 192;
    hipMemcpyAsync(outF + OFF_F0, h, (size_t)rows * D * sizeof(float),
                   hipMemcpyDeviceToDevice, stream);
  }

  // ================= stage 1 =================
  for (int i = 0; i < 2; ++i) run_layer(h, rows, H * W, D, layers1[i]);
  {
    int tmge = 16 * (H / 2) * (W / 2) * 4 * D;
    k_merge<<<blk(tmge), 256, 0, stream>>>(h, xz, H, W, D, tmge);
    ln(xz, 4 * D, m1g, m1b, mlp1, 4 * D, rows / 4, 4 * D);
    gemm(mlp1, 4 * D, m1w, 384, nullptr, h, 384, rows / 4, 384, 4 * D, 0, 0);
    rows /= 4; H /= 2; W /= 2; D = 384;
    hipMemcpyAsync(outF + OFF_F1, h, (size_t)rows * D * sizeof(float),
                   hipMemcpyDeviceToDevice, stream);
  }

  // ================= stage 2 =================
  for (int i = 0; i < 4; ++i) run_layer(h, rows, H * W, D, layers2[i]);
  hipMemcpyAsync(outF + OFF_F2, h, (size_t)rows * D * sizeof(float),
                 hipMemcpyDeviceToDevice, stream);

  // final norm -> out slot 0
  ln(h, D, fng, fnb, outF, D, rows, D);
}
